// DifferentialAttention_7267084665210
// MI455X (gfx1250) — matrix operand order repaired and checked by reading, not compiled, not run on hardware
//
#include <hip/hip_runtime.h>
#include <hip/hip_bf16.h>

typedef unsigned short u16;
typedef __attribute__((ext_vector_type(16))) __bf16 bf16x16;
typedef __attribute__((ext_vector_type(8)))  float  f32x8;
typedef __attribute__((ext_vector_type(4)))  unsigned int u32x4;
typedef __attribute__((ext_vector_type(8)))  int i32x8;
typedef __attribute__((ext_vector_type(4)))  int i32x4;

#define S_LEN 2048
#define D_MODEL 2048
#define NHEAD 16
#define BATCH 2
#define BH (BATCH * NHEAD)      // 32
#define MROWS (BATCH * S_LEN)   // 4096
#define NQKV 6144
#define ATT_SCALE 0.125f        // 64^-0.5
#define OUT_SCALE 0.2f          // 1 - LBDA_INIT

#if defined(__has_builtin)
#  if __has_builtin(__builtin_amdgcn_tensor_load_to_lds)
#    define HAS_TDM 1
#  endif
#endif
#ifndef HAS_TDM
#  define HAS_TDM 0
#endif

union Frag { uint4 q[2]; bf16x16 v; };

__device__ __forceinline__ u16 f2bf(float f) {
    unsigned int u = __float_as_uint(f);
    u += 0x7fffu + ((u >> 16) & 1u);
    return (u16)(u >> 16);
}
__device__ __forceinline__ float bf2f(u16 h) {
    return __uint_as_float(((unsigned int)h) << 16);
}
__device__ __forceinline__ uint4 ldg4(const u16* p) {
    return *(const uint4*)p;
}
__device__ __forceinline__ f32x8 zero8() {
    f32x8 z = {0.f,0.f,0.f,0.f,0.f,0.f,0.f,0.f};
    return z;
}
__device__ __forceinline__ f32x8 wmma_bf16(bf16x16 a, bf16x16 b, f32x8 c) {
    return __builtin_amdgcn_wmma_f32_16x16x32_bf16(false, a, false, b, (short)0, c, false, false);
}
__device__ __forceinline__ float half_max(float v) {
    v = fmaxf(v, __shfl_xor(v, 1));
    v = fmaxf(v, __shfl_xor(v, 2));
    v = fmaxf(v, __shfl_xor(v, 4));
    v = fmaxf(v, __shfl_xor(v, 8));
    return v;
}
__device__ __forceinline__ float half_sum(float v) {
    v += __shfl_xor(v, 1);
    v += __shfl_xor(v, 2);
    v += __shfl_xor(v, 4);
    v += __shfl_xor(v, 8);
    return v;
}

#if HAS_TDM
// Issue a 2D tensor DMA: global (row-major, elem=2B) tile -> LDS (packed rows).
// tile_d0 = contiguous extent (elements), tile_d1 = #rows, stride0 = row pitch (elements).
// amdgpu-toolchain (clang-23) 6-arg form:
//   (uint32x4 g0, int32x8 g1, int32x4 g2, int32x4 g3, int32x8 pad, i32 cpol)
__device__ __forceinline__ void tdm_load_2d(unsigned lds_off, const void* gaddr,
                                            unsigned tensor_d0, unsigned tensor_d1,
                                            unsigned long long stride0,
                                            unsigned tile_d0, unsigned tile_d1) {
    unsigned long long ga = (unsigned long long)gaddr;
    u32x4 g0;
    g0[0] = 1u;                                   // count=1 valid descriptor
    g0[1] = lds_off;                              // LDS byte address
    g0[2] = (unsigned)ga;                         // global addr [31:0]
    g0[3] = (unsigned)((ga >> 32) & 0x01FFFFFFu) | (2u << 30);  // addr[56:32] | type=2
    i32x8 g1;
    g1[0] = (int)(1u << 16);                      // data_size=1 (2 bytes), wg_mask=0
    g1[1] = (int)((tensor_d0 & 0xFFFFu) << 16);   // tensor_dim0[15:0]
    g1[2] = (int)((tensor_d0 >> 16) | ((tensor_d1 & 0xFFFFu) << 16));
    g1[3] = (int)((tensor_d1 >> 16) | (tile_d0 << 16));
    g1[4] = (int)(tile_d1 & 0xFFFFu);             // tile_dim1, tile_dim2=0
    g1[5] = (int)(unsigned)stride0;               // tensor_dim0_stride[31:0]
    g1[6] = (int)((unsigned)(stride0 >> 32) & 0xFFFFu);
    g1[7] = 0;
    i32x4 zz4 = {0, 0, 0, 0};
    i32x8 zz8 = {0, 0, 0, 0, 0, 0, 0, 0};
    __builtin_amdgcn_tensor_load_to_lds(g0, g1, zz4, zz4, zz8, 0);
}
#endif

__device__ __forceinline__ unsigned lds_offset_of(const void* p) {
    return (unsigned)(unsigned long long)p;       // low 32 bits = LDS byte offset
}

// ---------------- fp32 -> bf16 elementwise convert ----------------
__global__ __launch_bounds__(256) void k_cvt_bf16(const float* __restrict__ src,
                                                  u16* __restrict__ dst, int n4) {
    int i = blockIdx.x * blockDim.x + threadIdx.x;
    if (i < n4) {
        float4 v = ((const float4*)src)[i];
        ushort4 o;
        o.x = f2bf(v.x); o.y = f2bf(v.y); o.z = f2bf(v.z); o.w = f2bf(v.w);
        ((ushort4*)dst)[i] = o;
    }
}

// ---------------- fp32 (R x C) -> bf16 transposed (C x R) ----------------
__global__ __launch_bounds__(256) void k_transpose_bf16(const float* __restrict__ W,
                                                        u16* __restrict__ WT,
                                                        int R, int C) {
    __shared__ u16 tile[32][33];
    int c0 = blockIdx.x * 32, r0 = blockIdx.y * 32;
    for (int i = threadIdx.y; i < 32; i += 8) {
        tile[i][threadIdx.x] = f2bf(W[(size_t)(r0 + i) * C + (c0 + threadIdx.x)]);
    }
    __syncthreads();
    for (int i = threadIdx.y; i < 32; i += 8) {
        WT[(size_t)(c0 + i) * R + (r0 + threadIdx.x)] = tile[threadIdx.x][i];
    }
}

// =====================================================================
// Cooperative GEMM core: 256 threads = 8 waves, block tile 64(M) x 256(N),
// wave tile 32x64, K stepped by 32.  B tile (256 x 32 bf16) staged in LDS,
// double-buffered via Tensor Data Mover (wave 0 issues DMA one step ahead).
// =====================================================================
#define GEMM_PROLOGUE(Aptr, Btptr, Kdim, Ndim)                                   \
    __shared__ __align__(16) u16 bs[2][256 * 32];                                \
    const int K = (Kdim);                                                        \
    int tid = threadIdx.x;                                                       \
    int wv = tid >> 5;                                                           \
    int lane = tid & 31;                                                         \
    int mw = wv & 1;                                                             \
    int nw = wv >> 1;                                                            \
    int bm = blockIdx.x & 63;            /* MROWS/64 = 64 m-blocks */            \
    int bn = blockIdx.x >> 6;                                                    \
    int m0 = bm * 64 + mw * 32;                                                  \
    int nblk0 = bn * 256;                                                        \
    int n0 = nblk0 + nw * 64;                                                    \
    int nl0 = nw * 64;                                                           \
    int rlo = lane & 15;                                                         \
    int kbA = (lane >> 4) * 16;                                                  \
    int kbB = (lane >> 4) * 16;                                                  \
    int mhi = (lane >> 4) << 3;                                                  \
    const u16* pa0 = (Aptr) + (size_t)(m0 + rlo) * K;                            \
    const u16* pa1 = (Aptr) + (size_t)(m0 + 16 + rlo) * K;                       \
    const u16* btile = (Btptr) + (size_t)nblk0 * K;                              \
    f32x8 acc[2][4];                                                             \
    _Pragma("unroll") for (int i = 0; i < 2; ++i)                                \
        _Pragma("unroll") for (int j = 0; j < 4; ++j) acc[i][j] = zero8();

#if HAS_TDM
#define GEMM_STAGE_FIRST()                                                       \
    if (wv == 0)                                                                 \
        tdm_load_2d(lds_offset_of(&bs[0][0]), btile, K, 256, K, 32, 256);
#define GEMM_STAGE_STEP(kt, KT)                                                  \
    if (wv == 0) __builtin_amdgcn_s_wait_tensorcnt(0);                           \
    __syncthreads();                                                             \
    if (wv == 0 && (kt) + 1 < (KT))                                              \
        tdm_load_2d(lds_offset_of(&bs[((kt) + 1) & 1][0]),                       \
                    btile + ((kt) + 1) * 32, K, 256, K, 32, 256);
#else
#define GEMM_STAGE_FIRST()
#define GEMM_STAGE_STEP(kt, KT)                                                  \
    __syncthreads();                                                             \
    {                                                                            \
        const uint4* s4 = (const uint4*)(btile + (size_t)tid * K + (kt) * 32);   \
        uint4* d4 = (uint4*)&bs[(kt) & 1][tid * 32];                             \
        d4[0] = s4[0]; d4[1] = s4[1]; d4[2] = s4[2]; d4[3] = s4[3];              \
    }                                                                            \
    __syncthreads();
#endif

#define GEMM_COMPUTE_STEP(kt)                                                    \
    {                                                                            \
        const u16* bb = &bs[(kt) & 1][0];                                        \
        int k0 = (kt) * 32;                                                      \
        __builtin_prefetch(pa0 + k0 + 256, 0, 1);                                \
        Frag a[2], b[4];                                                         \
        a[0].q[0] = ldg4(pa0 + k0 + kbA); a[0].q[1] = ldg4(pa0 + k0 + kbA + 8);  \
        a[1].q[0] = ldg4(pa1 + k0 + kbA); a[1].q[1] = ldg4(pa1 + k0 + kbA + 8);  \
        _Pragma("unroll") for (int j = 0; j < 4; ++j) {                          \
            const u16* pb = bb + (size_t)(nl0 + j * 16 + rlo) * 32;              \
            b[j].q[0] = *(const uint4*)(pb + kbB);                               \
            b[j].q[1] = *(const uint4*)(pb + kbB + 8);                           \
        }                                                                        \
        _Pragma("unroll") for (int i = 0; i < 2; ++i)                            \
            _Pragma("unroll") for (int j = 0; j < 4; ++j)                        \
                acc[i][j] = wmma_bf16(a[i].v, b[j].v, acc[i][j]);                \
    }

// ---------------- QKV GEMM: (4096x2048) x (2048x6144), scatter epilogue ----------------
__global__ __launch_bounds__(256) void k_gemm_qkv(const u16* __restrict__ A,
                                                  const u16* __restrict__ Bt,
                                                  u16* __restrict__ qb,
                                                  u16* __restrict__ kbuf,
                                                  u16* __restrict__ vT) {
    GEMM_PROLOGUE(A, Bt, D_MODEL, NQKV)
    GEMM_STAGE_FIRST()
    const int KT = D_MODEL / 32;
    for (int kt = 0; kt < KT; ++kt) {
        GEMM_STAGE_STEP(kt, KT)
        GEMM_COMPUTE_STEP(kt)
    }

#pragma unroll
    for (int i = 0; i < 2; ++i)
#pragma unroll
        for (int j = 0; j < 4; ++j)
#pragma unroll
            for (int r = 0; r < 8; ++r) {
                int m = m0 + i * 16 + r + mhi;
                int n = n0 + j * 16 + rlo;
                u16 o = f2bf(acc[i][j][r]);
                int which = n >> 11;
                int rem = n & 2047;
                int h = rem >> 7;
                int d = rem & 127;
                int b_ = m >> 11;
                int s = m & 2047;
                int bh = b_ * NHEAD + h;
                if (which == 0)      qb  [((size_t)bh * S_LEN + s) * 128 + d] = o;
                else if (which == 1) kbuf[((size_t)bh * S_LEN + s) * 128 + d] = o;
                else                 vT  [((size_t)bh * 128 + d) * S_LEN + s] = o;
            }
}

// ---------------- Output GEMM: (4096x2048) x (2048x2048) -> fp32 ----------------
__global__ __launch_bounds__(256) void k_gemm_out(const u16* __restrict__ A,
                                                  const u16* __restrict__ Bt,
                                                  float* __restrict__ C) {
    GEMM_PROLOGUE(A, Bt, D_MODEL, D_MODEL)
    GEMM_STAGE_FIRST()
    const int KT = D_MODEL / 32;
    for (int kt = 0; kt < KT; ++kt) {
        GEMM_STAGE_STEP(kt, KT)
        GEMM_COMPUTE_STEP(kt)
    }

#pragma unroll
    for (int i = 0; i < 2; ++i)
#pragma unroll
        for (int j = 0; j < 4; ++j)
#pragma unroll
            for (int r = 0; r < 8; ++r) {
                int m = m0 + i * 16 + r + mhi;
                int n = n0 + j * 16 + rlo;
                C[(size_t)m * D_MODEL + n] = acc[i][j][r];
            }
}

// ---------------- RoPE in-place on q,k (bf16, (B,H,S,128)) ----------------
__global__ __launch_bounds__(256) void k_rope(u16* __restrict__ qb, u16* __restrict__ kbuf,
                                              const float* __restrict__ cosp,
                                              const float* __restrict__ sinp) {
    int i = blockIdx.x * blockDim.x + threadIdx.x;   // over BH * S * 32
    int p = i & 31;
    int s = (i >> 5) & (S_LEN - 1);
    int bh = i >> 16;                                // 2048*32 = 65536 per bh
    if (bh >= BH) return;
    float c  = cosp[s * 32 + p];
    float sn = sinp[s * 32 + p];
    size_t base = ((size_t)bh * S_LEN + s) * 128;
#pragma unroll
    for (int hb = 0; hb < 128; hb += 64) {
        u16* pq = qb + base + hb + 2 * p;
        float x1 = bf2f(pq[0]), x2 = bf2f(pq[1]);
        pq[0] = f2bf(x1 * c - x2 * sn);
        pq[1] = f2bf(x1 * sn + x2 * c);
        u16* pk = kbuf + base + hb + 2 * p;
        float y1 = bf2f(pk[0]), y2 = bf2f(pk[1]);
        pk[0] = f2bf(y1 * c - y2 * sn);
        pk[1] = f2bf(y1 * sn + y2 * c);
    }
}

// ---------------- online softmax update helper ----------------
__device__ __forceinline__ void online_update(const f32x8* sblk, int qbase, int mhi, int rlo,
                                              int ks0, float* m_, float* l_, float* c_,
                                              u16 (*ps)[32]) {
#pragma unroll
    for (int r = 0; r < 8; ++r) {
        int sq = qbase + r + mhi;
        float v0 = sblk[0][r] * ATT_SCALE;
        float v1 = sblk[1][r] * ATT_SCALE;
        if (ks0 + rlo > sq)      v0 = -1e30f;
        if (ks0 + 16 + rlo > sq) v1 = -1e30f;
        float tm = half_max(fmaxf(v0, v1));
        float mn = fmaxf(m_[r], tm);
        float corr = __expf(m_[r] - mn);
        float e0 = __expf(v0 - mn);
        float e1 = __expf(v1 - mn);
        float rs = half_sum(e0 + e1);
        l_[r] = l_[r] * corr + rs;
        m_[r] = mn;
        c_[r] = corr;
        ps[r + mhi][rlo]      = f2bf(e0);
        ps[r + mhi][16 + rlo] = f2bf(e1);
    }
}

// ---------------- differential flash attention: 1 wave per 16-row q tile ----------------
__global__ __launch_bounds__(32) void k_attn(const u16* __restrict__ qb,
                                             const u16* __restrict__ kbuf,
                                             const u16* __restrict__ vT,
                                             const float* __restrict__ gamma,
                                             const float* __restrict__ beta,
                                             const float* __restrict__ lbda_p,
                                             u16* __restrict__ attn_out) {
    __shared__ __align__(16) u16 p1s[16][32];
    __shared__ __align__(16) u16 p2s[16][32];

    int wid = blockIdx.x;                 // 0 .. BH*(S/16)-1
    int qt = wid % (S_LEN / 16);
    int bh = wid / (S_LEN / 16);
    int h = bh % NHEAD;
    int b = bh / NHEAD;
    int lane = threadIdx.x;
    int rlo = lane & 15;
    int kbA = (lane >> 4) * 16;
    int kbB = (lane >> 4) * 16;
    int mhi = (lane >> 4) << 3;
    const int qbase = qt * 16;
    float lb = lbda_p[0];

    const u16* pq = qb + ((size_t)bh * S_LEN + qbase + rlo) * 128;
    Frag q1a[2], q2a[2];
#pragma unroll
    for (int d = 0; d < 2; ++d) {
        q1a[d].q[0] = ldg4(pq + d * 32 + kbA);
        q1a[d].q[1] = ldg4(pq + d * 32 + kbA + 8);
        q2a[d].q[0] = ldg4(pq + 64 + d * 32 + kbA);
        q2a[d].q[1] = ldg4(pq + 64 + d * 32 + kbA + 8);
    }

    f32x8 o1[8], o2[8];
#pragma unroll
    for (int db = 0; db < 8; ++db) { o1[db] = zero8(); o2[db] = zero8(); }
    float m1[8], l1[8], c1[8], m2[8], l2[8], c2[8];
#pragma unroll
    for (int r = 0; r < 8; ++r) {
        m1[r] = -1e30f; l1[r] = 0.f; m2[r] = -1e30f; l2[r] = 0.f;
    }

    int ktiles = (qbase + 16 + 31) >> 5;
    for (int kt = 0; kt < ktiles; ++kt) {
        int ks0 = kt * 32;
        f32x8 s1[2], s2[2];
#pragma unroll
        for (int jb = 0; jb < 2; ++jb) {
            const u16* pk = kbuf + ((size_t)bh * S_LEN + ks0 + jb * 16 + rlo) * 128;
            Frag kf0, kf1, kg0, kg1;
            kf0.q[0] = ldg4(pk + kbB);       kf0.q[1] = ldg4(pk + kbB + 8);
            kf1.q[0] = ldg4(pk + 32 + kbB);  kf1.q[1] = ldg4(pk + 32 + kbB + 8);
            kg0.q[0] = ldg4(pk + 64 + kbB);  kg0.q[1] = ldg4(pk + 64 + kbB + 8);
            kg1.q[0] = ldg4(pk + 96 + kbB);  kg1.q[1] = ldg4(pk + 96 + kbB + 8);
            f32x8 a1 = zero8();
            a1 = wmma_bf16(q1a[0].v, kf0.v, a1);
            a1 = wmma_bf16(q1a[1].v, kf1.v, a1);
            s1[jb] = a1;
            f32x8 a2 = zero8();
            a2 = wmma_bf16(q2a[0].v, kg0.v, a2);
            a2 = wmma_bf16(q2a[1].v, kg1.v, a2);
            s2[jb] = a2;
        }

        online_update(s1, qbase, mhi, rlo, ks0, m1, l1, c1, p1s);
        online_update(s2, qbase, mhi, rlo, ks0, m2, l2, c2, p2s);
        __syncthreads();   // single-wave WG: NOP barrier + LDS ordering fence

        Frag p1a, p2a;
        p1a.q[0] = *(const uint4*)&p1s[rlo][kbA];
        p1a.q[1] = *(const uint4*)&p1s[rlo][kbA + 8];
        p2a.q[0] = *(const uint4*)&p2s[rlo][kbA];
        p2a.q[1] = *(const uint4*)&p2s[rlo][kbA + 8];

#pragma unroll
        for (int db = 0; db < 8; ++db) {
            const u16* pv = vT + ((size_t)bh * 128 + db * 16 + rlo) * S_LEN + ks0;
            Frag vb;
            vb.q[0] = ldg4(pv + kbB);
            vb.q[1] = ldg4(pv + kbB + 8);
            f32x8 t1 = o1[db], t2 = o2[db];
#pragma unroll
            for (int r = 0; r < 8; ++r) { t1[r] *= c1[r]; t2[r] *= c2[r]; }
            o1[db] = wmma_bf16(p1a.v, vb.v, t1);
            o2[db] = wmma_bf16(p2a.v, vb.v, t2);
        }
        __syncthreads();
    }

    float a1r[8], a2r[8];
#pragma unroll
    for (int r = 0; r < 8; ++r) {
        a1r[r] = 1.f / l1[r];
        a2r[r] = lb / l2[r];
    }
#pragma unroll
    for (int db = 0; db < 8; ++db)
#pragma unroll
        for (int r = 0; r < 8; ++r)
            o1[db][r] = o1[db][r] * a1r[r] - o2[db][r] * a2r[r];

    float mu[8], rstd[8];
#pragma unroll
    for (int r = 0; r < 8; ++r) {
        float s = 0.f, ss = 0.f;
#pragma unroll
        for (int db = 0; db < 8; ++db) {
            float v = o1[db][r];
            s += v; ss += v * v;
        }
        s = half_sum(s);
        ss = half_sum(ss);
        float mean = s * (1.f / 128.f);
        float var = ss * (1.f / 128.f) - mean * mean;
        mu[r] = mean;
        rstd[r] = rsqrtf(var + 1e-5f);
    }
#pragma unroll
    for (int db = 0; db < 8; ++db) {
        int d = db * 16 + rlo;
        float g  = gamma[h * 128 + d];
        float be = beta[h * 128 + d];
#pragma unroll
        for (int r = 0; r < 8; ++r) {
            float v = ((o1[db][r] - mu[r]) * rstd[r] * g + be) * OUT_SCALE;
            size_t dst = ((size_t)b * S_LEN + qbase + r + mhi) * (size_t)D_MODEL + h * 128 + d;
            attn_out[dst] = f2bf(v);
        }
    }
}

extern "C" void kernel_launch(void* const* d_in, const int* in_sizes, int n_in,
                              void* d_out, int out_size, void* d_ws, size_t ws_size,
                              hipStream_t stream) {
    const float* x     = (const float*)d_in[0];
    // d_in[1] = mask (causal -1e9), implemented directly in k_attn
    const float* cosp  = (const float*)d_in[2];
    const float* sinp  = (const float*)d_in[3];
    const float* Wqkv  = (const float*)d_in[4];
    const float* Wo    = (const float*)d_in[5];
    const float* gamma = (const float*)d_in[6];
    const float* beta  = (const float*)d_in[7];
    const float* lbda  = (const float*)d_in[8];

    unsigned char* wsp = (unsigned char*)d_ws;
    auto take = [&](size_t bytes) -> void* {
        void* p = (void*)wsp;
        wsp += (bytes + 255) & ~(size_t)255;
        return p;
    };
    u16* xb  = (u16*)take((size_t)MROWS * D_MODEL * 2);      // x in bf16
    u16* WqT = (u16*)take((size_t)NQKV * D_MODEL * 2);       // W_qkv^T bf16
    u16* WoT = (u16*)take((size_t)D_MODEL * D_MODEL * 2);    // W_o^T bf16
    u16* qb  = (u16*)take((size_t)BH * S_LEN * 128 * 2);     // q (B,H,S,128)
    u16* kb  = (u16*)take((size_t)BH * S_LEN * 128 * 2);     // k (B,H,S,128)
    u16* vT  = (u16*)take((size_t)BH * S_LEN * 128 * 2);     // v^T (B,H,128,S)
    u16* at  = (u16*)take((size_t)MROWS * D_MODEL * 2);      // attn out (B,S,D) bf16

    k_cvt_bf16<<<(MROWS * D_MODEL / 4) / 256, 256, 0, stream>>>(x, xb, MROWS * D_MODEL / 4);
    k_transpose_bf16<<<dim3(NQKV / 32, D_MODEL / 32), dim3(32, 8), 0, stream>>>(Wqkv, WqT, D_MODEL, NQKV);
    k_transpose_bf16<<<dim3(D_MODEL / 32, D_MODEL / 32), dim3(32, 8), 0, stream>>>(Wo, WoT, D_MODEL, D_MODEL);
    k_gemm_qkv<<<64 * (NQKV / 256), 256, 0, stream>>>(xb, WqT, qb, kb, vT);
    k_rope<<<(BH * S_LEN * 32) / 256, 256, 0, stream>>>(qb, kb, cosp, sinp);
    k_attn<<<BH * (S_LEN / 16), 32, 0, stream>>>(qb, kb, vT, gamma, beta, lbda, at);
    k_gemm_out<<<64 * (D_MODEL / 256), 256, 0, stream>>>(at, WoT, (float*)d_out);
}
